// StableHypersphericalPrototype_6064493822098
// MI455X (gfx1250) — matrix operand order repaired and checked by reading, not compiled, not run on hardware
//
#include <hip/hip_runtime.h>

typedef __bf16 bf16;
typedef __attribute__((ext_vector_type(16))) __bf16 v16bf;
typedef __attribute__((ext_vector_type(8)))  float  v8f;

#define BSZ 16384
#define FDIM 1024
#define HDIM 512
#define PDIM 64
#define DDIM 8
#define DP   512   // D*P

// ---------------------------------------------------------------------------
// Register-blocked bf16 WMMA GEMM: C[M,N] = A[M,K] @ Bt[N,K]^T + bias
// Block: 256 threads = 8 waves arranged 4(M) x 2(N). Wave tile 32x64,
// block tile 128x128. Requires M%128==0, N%128==0, K%32==0.
// Per K-step per wave: 2 A-frags + 4 B-frags (12 x b128) -> 8 x v_wmma.
// ---------------------------------------------------------------------------
__global__ void gemm_bf16_wmma_big(const bf16* __restrict__ A,
                                   const bf16* __restrict__ Bt,
                                   const float* __restrict__ bias,
                                   float* __restrict__ C,
                                   int M, int N, int K)
{
    const int wave = threadIdx.x >> 5;
    const int lane = threadIdx.x & 31;
    const int half = lane >> 4;     // 0: lanes 0-15, 1: lanes 16-31
    const int l16  = lane & 15;
    const int wm   = wave >> 1;     // 0..3
    const int wn   = wave & 1;      // 0..1

    const int m0 = blockIdx.y * 128 + wm * 32;
    const int n0 = blockIdx.x * 128 + wn * 64;

    v8f acc[2][4] = {};
    for (int k0 = 0; k0 < K; k0 += 32) {
        // A fragments (16x32 bf16 each): lane<16 -> K = k0..+15
        //                                lane>=16 -> K = k0+16..+31
        v16bf a[2];
#pragma unroll
        for (int im = 0; im < 2; ++im) {
            const bf16* pa = A + (size_t)(m0 + im * 16 + l16) * K + k0 + half * 16;
            if (k0 + 64 < K) __builtin_prefetch(pa + 32, 0, 1);
#pragma unroll
            for (int i = 0; i < 8; ++i) a[im][i] = pa[i];
#pragma unroll
            for (int i = 0; i < 8; ++i) a[im][8 + i] = pa[8 + i];
        }
        // B fragments (32x16 bf16): lane column, 16 contiguous K values
        v16bf b[4];
#pragma unroll
        for (int in = 0; in < 4; ++in) {
            const bf16* pb = Bt + (size_t)(n0 + in * 16 + l16) * K + k0 + half * 16;
#pragma unroll
            for (int i = 0; i < 16; ++i) b[in][i] = pb[i];
        }
#pragma unroll
        for (int im = 0; im < 2; ++im)
#pragma unroll
            for (int in = 0; in < 4; ++in)
                acc[im][in] = __builtin_amdgcn_wmma_f32_16x16x32_bf16(
                    false, a[im], false, b[in], (short)0, acc[im][in], false, false);
    }
    // C/D layout: VGPR r -> M = tile_m + half*8 + r, N = tile_n + l16
#pragma unroll
    for (int im = 0; im < 2; ++im) {
#pragma unroll
        for (int in = 0; in < 4; ++in) {
            const int nn = n0 + in * 16 + l16;
            const float bv = bias ? bias[nn] : 0.0f;
#pragma unroll
            for (int r = 0; r < 8; ++r) {
                const int mm = m0 + im * 16 + half * 8 + r;
                C[(size_t)mm * N + nn] = acc[im][in][r] + bv;
            }
        }
    }
}

// ---------------------------------------------------------------------------
// Small-N fallback (used for the [B,512]x[512,64] logits GEMM).
// Block 256 = 8 waves; block tile 32(M) x 64(N); wave tile 16x16.
// ---------------------------------------------------------------------------
__global__ void gemm_bf16_wmma(const bf16* __restrict__ A,
                               const bf16* __restrict__ Bt,
                               const float* __restrict__ bias,
                               float* __restrict__ C,
                               int M, int N, int K)
{
    const int wave = threadIdx.x >> 5;
    const int lane = threadIdx.x & 31;
    const int half = lane >> 4;
    const int l16  = lane & 15;

    const int m0 = blockIdx.y * 32 + (wave & 1) * 16;
    const int n0 = blockIdx.x * 64 + (wave >> 1) * 16;

    v8f acc = {};
    for (int k0 = 0; k0 < K; k0 += 32) {
        v16bf a;
        {
            const bf16* pa = A + (size_t)(m0 + l16) * K + k0 + half * 16;
#pragma unroll
            for (int i = 0; i < 8; ++i) a[i] = pa[i];
#pragma unroll
            for (int i = 0; i < 8; ++i) a[8 + i] = pa[8 + i];
        }
        v16bf b;
        {
            const bf16* pb = Bt + (size_t)(n0 + l16) * K + k0 + half * 16;
#pragma unroll
            for (int i = 0; i < 16; ++i) b[i] = pb[i];
        }
        acc = __builtin_amdgcn_wmma_f32_16x16x32_bf16(
            false, a, false, b, (short)0, acc, false, false);
    }
#pragma unroll
    for (int r = 0; r < 8; ++r) {
        const int mm = m0 + half * 8 + r;
        const int nn = n0 + l16;
        float v = acc[r];
        if (bias) v += bias[nn];
        C[(size_t)mm * N + nn] = v;
    }
}

// ---------------------------------------------------------------------------
// Elementwise / reduction helpers
// ---------------------------------------------------------------------------
__global__ void cast_bf16_kernel(const float* __restrict__ in,
                                 bf16* __restrict__ out, size_t n)
{
    size_t i = (size_t)blockIdx.x * 256 + threadIdx.x;
    if (i < n) out[i] = (bf16)in[i];
}

// in: [K,N] fp32 row-major -> out: [N,K] bf16
__global__ void transpose_cast_kernel(const float* __restrict__ in,
                                      bf16* __restrict__ out, int K, int N)
{
    size_t i = (size_t)blockIdx.x * 256 + threadIdx.x;
    if (i >= (size_t)K * N) return;
    int n = (int)(i / K);
    int k = (int)(i % K);
    out[i] = (bf16)in[(size_t)k * N + n];
}

__global__ void zero_f32_kernel(float* __restrict__ p, size_t n)
{
    size_t i = (size_t)blockIdx.x * 256 + threadIdx.x;
    if (i < n) p[i] = 0.0f;
}

// Row LayerNorm + exact GELU, write bf16 (next GEMM's A operand). 1 block/row.
__global__ void ln_gelu_kernel(const float* __restrict__ X,
                               const float* __restrict__ g,
                               const float* __restrict__ be,
                               bf16* __restrict__ Y, int L)
{
    const int row = blockIdx.x;
    const float* x = X + (size_t)row * L;
    __shared__ float red[256];
    float s = 0.f, s2 = 0.f;
    for (int i = threadIdx.x; i < L; i += 256) { float v = x[i]; s += v; s2 += v * v; }
    red[threadIdx.x] = s; __syncthreads();
    for (int st = 128; st > 0; st >>= 1) {
        if (threadIdx.x < st) red[threadIdx.x] += red[threadIdx.x + st];
        __syncthreads();
    }
    const float mean = red[0] / (float)L; __syncthreads();
    red[threadIdx.x] = s2; __syncthreads();
    for (int st = 128; st > 0; st >>= 1) {
        if (threadIdx.x < st) red[threadIdx.x] += red[threadIdx.x + st];
        __syncthreads();
    }
    const float var = red[0] / (float)L - mean * mean;
    const float inv = rsqrtf(var + 1e-5f);
    for (int i = threadIdx.x; i < L; i += 256) {
        float v = (x[i] - mean) * inv * g[i] + be[i];
        float ge = 0.5f * v * (1.0f + erff(v * 0.70710678118654752f));
        Y[(size_t)row * L + i] = (bf16)ge;
    }
}

// Row L2 normalize; write fp32 + bf16 copies. 1 block/row.
__global__ void l2norm_kernel(const float* __restrict__ X,
                              float* __restrict__ Yf,
                              bf16* __restrict__ Yb, int L)
{
    const int row = blockIdx.x;
    const float* x = X + (size_t)row * L;
    __shared__ float red[256];
    float ss = 0.f;
    for (int i = threadIdx.x; i < L; i += 256) { float v = x[i]; ss += v * v; }
    red[threadIdx.x] = ss; __syncthreads();
    for (int st = 128; st > 0; st >>= 1) {
        if (threadIdx.x < st) red[threadIdx.x] += red[threadIdx.x + st];
        __syncthreads();
    }
    const float inv = 1.0f / fmaxf(sqrtf(red[0]), 1e-12f);
    for (int i = threadIdx.x; i < L; i += 256) {
        float v = x[i] * inv;
        Yf[(size_t)row * L + i] = v;
        Yb[(size_t)row * L + i] = (bf16)v;
    }
}

// Softmax over 64 logits per row; 64 threads/block.
__global__ void softmax64_kernel(const float* __restrict__ logits,
                                 float* __restrict__ out)
{
    const int row = blockIdx.x;
    const int t = threadIdx.x;
    __shared__ float sh[64];
    float v = logits[(size_t)row * 64 + t];
    sh[t] = v; __syncthreads();
    for (int st = 32; st > 0; st >>= 1) {
        if (t < st) sh[t] = fmaxf(sh[t], sh[t + st]);
        __syncthreads();
    }
    const float mx = sh[0]; __syncthreads();
    float e = expf(v - mx);
    sh[t] = e; __syncthreads();
    for (int st = 32; st > 0; st >>= 1) {
        if (t < st) sh[t] += sh[t + st];
        __syncthreads();
    }
    out[(size_t)row * 64 + t] = e / sh[0];
}

// Per sample: argmax of own-domain sims slice, then atomic segment-sum of feats.
__global__ void assign_scatter_kernel(const float* __restrict__ sims,   // [B, 512]
                                      const int* __restrict__ dom,      // [B]
                                      const float* __restrict__ feats,  // [B, F]
                                      float* __restrict__ sums,         // [512, F]
                                      float* __restrict__ cnts)         // [512]
{
    const int row = blockIdx.x;
    const int t = threadIdx.x;
    int d = dom[row]; d = d < 0 ? 0 : (d > DDIM - 1 ? DDIM - 1 : d);
    __shared__ float sv[64];
    __shared__ int   si[64];
    if (t < 64) { sv[t] = sims[(size_t)row * DP + d * PDIM + t]; si[t] = t; }
    __syncthreads();
    for (int st = 32; st > 0; st >>= 1) {
        if (t < st) {
            if (sv[t + st] > sv[t]) { sv[t] = sv[t + st]; si[t] = si[t + st]; }
        }
        __syncthreads();
    }
    const int seg = d * PDIM + si[0];
    for (int f = t; f < FDIM; f += 256)
        atomicAdd(&sums[(size_t)seg * FDIM + f], feats[(size_t)row * FDIM + f]);
    if (t == 0) atomicAdd(&cnts[seg], 1.0f);
}

// EMA prototype update, 1 block per (d,p) segment.
__global__ void update_protos_kernel(const float* __restrict__ sums,
                                     const float* __restrict__ cnts,
                                     const float* __restrict__ proto,  // [512,F]
                                     const float* __restrict__ uc,     // [512]
                                     float* __restrict__ upd,          // [512,F]
                                     float* __restrict__ ucnew)        // [512]
{
    const int seg = blockIdx.x;
    const int t = threadIdx.x;
    __shared__ float red[256];
    const float cnt = cnts[seg];
    const float denom = fmaxf(cnt, 1.0f);
    float ss = 0.f;
    for (int f = t; f < FDIM; f += 256) {
        float m = sums[(size_t)seg * FDIM + f] / denom;
        ss += m * m;
    }
    red[t] = ss; __syncthreads();
    for (int st = 128; st > 0; st >>= 1) {
        if (t < st) red[t] += red[t + st];
        __syncthreads();
    }
    const float inv = 1.0f / fmaxf(sqrtf(red[0]), 1e-12f);
    const bool  has = cnt > 0.0f;
    const float u = uc[seg];
    const float lr = powf(0.95f, 1.0f + u * 0.01f);
    for (int f = t; f < FDIM; f += 256) {
        float pv = proto[(size_t)seg * FDIM + f];
        float nv = (sums[(size_t)seg * FDIM + f] / denom) * inv;
        upd[(size_t)seg * FDIM + f] = has ? (lr * pv + (1.0f - lr) * nv) : pv;
    }
    if (t == 0) ucnew[seg] = u + (has ? 1.0f : 0.0f);
}

// Global prototype = l2norm(sum_d softmax(-uc)_d * upd[d,p]); also emit
// Mcomb^T[f, d*64+p] = bf16(0.6*upd + 0.4*glob) for the final GEMM.
__global__ void glob_mcomb_kernel(const float* __restrict__ ucnew,  // [512]
                                  const float* __restrict__ upd,    // [512,F]
                                  float* __restrict__ glob,         // [64,F]
                                  bf16* __restrict__ mcombT)        // [F,512]
{
    const int p = blockIdx.x;
    const int t = threadIdx.x;
    float gw[DDIM];
    float mx = -1e30f;
#pragma unroll
    for (int d = 0; d < DDIM; ++d) { gw[d] = -ucnew[d * PDIM + p]; mx = fmaxf(mx, gw[d]); }
    float s = 0.f;
#pragma unroll
    for (int d = 0; d < DDIM; ++d) { gw[d] = expf(gw[d] - mx); s += gw[d]; }
#pragma unroll
    for (int d = 0; d < DDIM; ++d) gw[d] /= s;

    __shared__ float red[256];
    float ss = 0.f;
    for (int f = t; f < FDIM; f += 256) {
        float gv = 0.f;
#pragma unroll
        for (int d = 0; d < DDIM; ++d) gv += gw[d] * upd[(size_t)(d * PDIM + p) * FDIM + f];
        glob[(size_t)p * FDIM + f] = gv;
        ss += gv * gv;
    }
    red[t] = ss; __syncthreads();
    for (int st = 128; st > 0; st >>= 1) {
        if (t < st) red[t] += red[t + st];
        __syncthreads();
    }
    const float inv = 1.0f / fmaxf(sqrtf(red[0]), 1e-12f);
    for (int f = t; f < FDIM; f += 256) {
        float gv = glob[(size_t)p * FDIM + f] * inv;
        glob[(size_t)p * FDIM + f] = gv;
#pragma unroll
        for (int d = 0; d < DDIM; ++d) {
            float m = 0.6f * upd[(size_t)(d * PDIM + p) * FDIM + f] + 0.4f * gv;
            mcombT[(size_t)f * DP + d * PDIM + p] = (bf16)m;
        }
    }
}

// pw_oh[b, d*64+p] = (d == dom[b]) ? proto_weights[b,p] : 0  (bf16)
__global__ void build_pwoh_kernel(const float* __restrict__ pw,  // [B,64]
                                  const int* __restrict__ dom,
                                  bf16* __restrict__ pwoh)       // [B,512]
{
    size_t i = (size_t)blockIdx.x * 256 + threadIdx.x;
    if (i >= (size_t)BSZ * DP) return;
    int b = (int)(i >> 9);
    int j = (int)(i & (DP - 1));
    int d = dom[b]; d = d < 0 ? 0 : (d > DDIM - 1 ? DDIM - 1 : d);
    float v = ((j >> 6) == d) ? pw[(size_t)b * PDIM + (j & 63)] : 0.0f;
    pwoh[i] = (bf16)v;
}

__global__ void final_add_kernel(const float* __restrict__ feats,
                                 const float* __restrict__ pfeat,
                                 float* __restrict__ out, size_t n)
{
    size_t i = (size_t)blockIdx.x * 256 + threadIdx.x;
    if (i < n) out[i] = feats[i] + 0.2f * pfeat[i];
}

// ---------------------------------------------------------------------------
extern "C" void kernel_launch(void* const* d_in, const int* in_sizes, int n_in,
                              void* d_out, int out_size, void* d_ws, size_t ws_size,
                              hipStream_t stream)
{
    const float* features = (const float*)d_in[0];
    const int*   dom      = (const int*)d_in[1];
    const float* w1  = (const float*)d_in[2];
    const float* b1  = (const float*)d_in[3];
    const float* g1  = (const float*)d_in[4];
    const float* be1 = (const float*)d_in[5];
    const float* w2  = (const float*)d_in[6];
    const float* b2  = (const float*)d_in[7];
    const float* pw1 = (const float*)d_in[8];
    const float* pb1 = (const float*)d_in[9];
    const float* pg1 = (const float*)d_in[10];
    const float* pbe1= (const float*)d_in[11];
    const float* pw2 = (const float*)d_in[12];
    const float* pb2 = (const float*)d_in[13];
    const float* dproto = (const float*)d_in[14];
    // d_in[15] global_prototypes: unused by the reference outputs
    const float* ucount = (const float*)d_in[16];

    char* ws = (char*)d_ws;
    size_t off = 0;
    auto alloc = [&](size_t bytes) -> void* {
        void* p = ws + off;
        off += (bytes + 255) & ~(size_t)255;
        return p;
    };

    bf16*  fa      = (bf16*) alloc((size_t)BSZ * FDIM * 2);   // features bf16
    bf16*  w1t     = (bf16*) alloc((size_t)FDIM * FDIM * 2);  // [F,F] transposed
    bf16*  w2t     = (bf16*) alloc((size_t)FDIM * FDIM * 2);
    bf16*  pw1t    = (bf16*) alloc((size_t)HDIM * FDIM * 2);  // [H,F]
    bf16*  pw2t    = (bf16*) alloc((size_t)PDIM * HDIM * 2);  // [P,H]
    bf16*  protoT  = (bf16*) alloc((size_t)DP * FDIM * 2);    // [512,F] (already N-major)
    float* y1      = (float*)alloc((size_t)BSZ * FDIM * 4);   // Y1 / Y2 / proto_feat
    bf16*  hb      = (bf16*) alloc((size_t)BSZ * FDIM * 2);   // gelu(ln(Y1)) bf16
    float* featsf  = (float*)alloc((size_t)BSZ * FDIM * 4);
    bf16*  featsb  = (bf16*) alloc((size_t)BSZ * FDIM * 2);
    float* yp      = (float*)alloc((size_t)BSZ * HDIM * 4);   // Yp, then sims
    bf16*  hpb     = (bf16*) alloc((size_t)BSZ * HDIM * 2);
    float* logits  = (float*)alloc((size_t)BSZ * PDIM * 4);
    float* sums    = (float*)alloc((size_t)DP * FDIM * 4);
    float* cnts    = (float*)alloc((size_t)DP * 4);
    float* ucnew   = (float*)alloc((size_t)DP * 4);
    float* updb    = (float*)alloc((size_t)DP * FDIM * 4);
    float* globb   = (float*)alloc((size_t)PDIM * FDIM * 4);
    bf16*  mcombT  = (bf16*) alloc((size_t)FDIM * DP * 2);    // [F,512]
    bf16*  pwoh    = (bf16*) alloc((size_t)BSZ * DP * 2);     // [B,512]

    float* out_enh = (float*)d_out;                   // [B,F]
    float* out_pw  = out_enh + (size_t)BSZ * FDIM;    // [B,P]

    const size_t nBF = (size_t)BSZ * FDIM;

    // --- precision prep: bf16 casts + weight transposes ---
    cast_bf16_kernel<<<(int)((nBF + 255) / 256), 256, 0, stream>>>(features, fa, nBF);
    transpose_cast_kernel<<<(int)(((size_t)FDIM * FDIM + 255) / 256), 256, 0, stream>>>(w1, w1t, FDIM, FDIM);
    transpose_cast_kernel<<<(int)(((size_t)FDIM * FDIM + 255) / 256), 256, 0, stream>>>(w2, w2t, FDIM, FDIM);
    transpose_cast_kernel<<<(int)(((size_t)FDIM * HDIM + 255) / 256), 256, 0, stream>>>(pw1, pw1t, FDIM, HDIM);
    transpose_cast_kernel<<<(int)(((size_t)HDIM * PDIM + 255) / 256), 256, 0, stream>>>(pw2, pw2t, HDIM, PDIM);
    cast_bf16_kernel<<<(int)(((size_t)DP * FDIM + 255) / 256), 256, 0, stream>>>(dproto, protoT, (size_t)DP * FDIM);
    zero_f32_kernel<<<(int)(((size_t)DP * FDIM + 255) / 256), 256, 0, stream>>>(sums, (size_t)DP * FDIM);
    zero_f32_kernel<<<(int)((DP + 255) / 256), 256, 0, stream>>>(cnts, DP);

    // --- projection head: feats = l2norm(gelu(ln(X@w1+b1)) @ w2 + b2) ---
    gemm_bf16_wmma_big<<<dim3(FDIM / 128, BSZ / 128), 256, 0, stream>>>(fa, w1t, b1, y1, BSZ, FDIM, FDIM);
    ln_gelu_kernel<<<BSZ, 256, 0, stream>>>(y1, g1, be1, hb, FDIM);
    gemm_bf16_wmma_big<<<dim3(FDIM / 128, BSZ / 128), 256, 0, stream>>>(hb, w2t, b2, y1, BSZ, FDIM, FDIM);
    l2norm_kernel<<<BSZ, 256, 0, stream>>>(y1, featsf, featsb, FDIM);

    // --- prototype-weight MLP: softmax(gelu(ln(X@pw1+pb1)) @ pw2 + pb2) ---
    gemm_bf16_wmma_big<<<dim3(HDIM / 128, BSZ / 128), 256, 0, stream>>>(fa, pw1t, pb1, yp, BSZ, HDIM, FDIM);
    ln_gelu_kernel<<<BSZ, 256, 0, stream>>>(yp, pg1, pbe1, hpb, HDIM);
    gemm_bf16_wmma<<<dim3(PDIM / 64, BSZ / 32), 256, 0, stream>>>(hpb, pw2t, pb2, logits, BSZ, PDIM, HDIM);
    softmax64_kernel<<<BSZ, 64, 0, stream>>>(logits, out_pw);

    // --- sims against all D*P prototypes, then assign + segment-sum ---
    float* sims = yp;  // yp consumed above; reuse
    gemm_bf16_wmma_big<<<dim3(DP / 128, BSZ / 128), 256, 0, stream>>>(featsb, protoT, nullptr, sims, BSZ, DP, FDIM);
    assign_scatter_kernel<<<BSZ, 256, 0, stream>>>(sims, dom, featsf, sums, cnts);

    // --- EMA update + global prototypes + combined matrix ---
    update_protos_kernel<<<DP, 256, 0, stream>>>(sums, cnts, dproto, ucount, updb, ucnew);
    glob_mcomb_kernel<<<PDIM, 256, 0, stream>>>(ucnew, updb, globb, mcombT);

    // --- enhancement: enhanced = feats + 0.2 * (pw_oh @ Mcomb) ---
    build_pwoh_kernel<<<(int)(((size_t)BSZ * DP + 255) / 256), 256, 0, stream>>>(out_pw, dom, pwoh);
    float* pfeat = y1;  // Y2 consumed; reuse
    gemm_bf16_wmma_big<<<dim3(FDIM / 128, BSZ / 128), 256, 0, stream>>>(pwoh, mcombT, nullptr, pfeat, BSZ, FDIM, DP);
    final_add_kernel<<<(int)((nBF + 255) / 256), 256, 0, stream>>>(featsf, pfeat, out_enh, nBF);
}
